// ODSTLayer_18081812316216
// MI455X (gfx1250) — matrix operand order repaired and checked by reading, not compiled, not run on hardware
//
#include <hip/hip_runtime.h>

typedef __bf16 v16bf __attribute__((ext_vector_type(16)));
typedef __bf16 v8bf  __attribute__((ext_vector_type(8)));
typedef float  v8f   __attribute__((ext_vector_type(8)));

#define B_DIM 4096
#define F_DIM 512
#define T_DIM 2048
#define D_DIM 4
#define TD    (T_DIM * D_DIM)          // 8192 rows of W
#define INV_TAU 4.9553888f             // 1 / 0.2018004745467103

// ---------------------------------------------------------------------------
// Kernel 1: row softmax over F=512 for each (t,d) row, emit bf16 weights.
// One wave per row (wave32), 16 elements per lane, shfl-xor reductions.
// ---------------------------------------------------------------------------
__global__ __launch_bounds__(256) void odst_softmax_bf16(
    const float* __restrict__ logits,   // (TD, F)
    __bf16* __restrict__ Wb)            // (TD, F)
{
    const int wave = threadIdx.x >> 5;
    const int lane = threadIdx.x & 31;
    const int row  = blockIdx.x * 8 + wave;

    const float* src = logits + (size_t)row * F_DIM;
    __bf16*      dst = Wb     + (size_t)row * F_DIM;

    float v[16];
    float m = -3.0e38f;
#pragma unroll
    for (int i = 0; i < 16; ++i) {
        v[i] = src[lane + i * 32];
        m = fmaxf(m, v[i]);
    }
#pragma unroll
    for (int off = 16; off >= 1; off >>= 1)
        m = fmaxf(m, __shfl_xor(m, off, 32));

    float s = 0.0f;
#pragma unroll
    for (int i = 0; i < 16; ++i) {
        v[i] = __expf(v[i] - m);
        s += v[i];
    }
#pragma unroll
    for (int off = 16; off >= 1; off >>= 1)
        s += __shfl_xor(s, off, 32);

    const float r = 1.0f / s;
#pragma unroll
    for (int i = 0; i < 16; ++i)
        dst[lane + i * 32] = (__bf16)(v[i] * r);
}

// ---------------------------------------------------------------------------
// Kernel 2: convert x (B,F) f32 -> bf16.
// ---------------------------------------------------------------------------
__global__ __launch_bounds__(256) void odst_x_bf16(
    const float* __restrict__ x, __bf16* __restrict__ Xb)
{
    const size_t i = (size_t)blockIdx.x * 256 + threadIdx.x;
    Xb[i] = (__bf16)x[i];
}

// ---------------------------------------------------------------------------
// Kernel 3: fused GEMM (bf16 WMMA, f32 acc) + sigmoid + 16-leaf reduction.
// Block = 256 threads = 8 waves. Workgroup tile: 128 B-rows x 64 td-cols.
// Each wave: 16 rows x 64 cols = 4 accumulators of v_wmma_f32_16x16x32_bf16,
// K-loop over F=512 in steps of 32. 64 cols == 16 complete trees, so the
// tree epilogue is tile-local: C -> LDS -> per-(b,t) sigmoid/leaf-sum -> out.
// ---------------------------------------------------------------------------
__global__ __launch_bounds__(256) void odst_fused(
    const __bf16* __restrict__ Xb,          // (B, F)
    const __bf16* __restrict__ Wb,          // (TD, F)
    const float*  __restrict__ thresholds,  // (T, D)
    const float*  __restrict__ leaf,        // (T, 16)
    float*        __restrict__ out)         // (B, T)
{
    __shared__ float xsel[8][16][68];       // per-wave 16x64 tile (+pad)

    const int wave = threadIdx.x >> 5;
    const int lane = threadIdx.x & 31;
    const int half = lane >> 4;             // 0: lanes 0-15, 1: lanes 16-31
    const int l16  = lane & 15;

    const int rowBase = blockIdx.x * 128 + wave * 16;  // B rows of this wave
    const int colBase = blockIdx.y * 64;               // td cols (16 trees)

    v8f acc[4] = {};                        // zero-initialized accumulators

    const __bf16* arow = Xb + (size_t)(rowBase + l16) * F_DIM;

    for (int kk = 0; kk < F_DIM; kk += 32) {
        // A fragment (16x32 bf16): lane M = l16,
        // half 0 holds K 0..15, half 1 holds K 16..31.
        v16bf a = *(const v16bf*)(arow + kk + half * 16);

#pragma unroll
        for (int n = 0; n < 4; ++n) {
            // B fragment (32x16): lane column N = l16 == row (colBase+n*16+l16)
            // of row-major W; half 0 holds K 0..15, half 1 holds K 16..31.
            const __bf16* brow =
                Wb + (size_t)(colBase + n * 16 + l16) * F_DIM + kk + half * 16;
            v16bf b = *(const v16bf*)brow;
            acc[n] = __builtin_amdgcn_wmma_f32_16x16x32_bf16(
                false, a, false, b, (short)0, acc[n], false, false);
        }
    }

    // C layout (7.12.2): VGPR j -> M = j + half*8, N = l16. Dump to LDS.
#pragma unroll
    for (int n = 0; n < 4; ++n)
#pragma unroll
        for (int j = 0; j < 8; ++j)
            xsel[wave][j + half * 8][n * 16 + l16] = acc[n][j];
    __syncthreads();

    // Epilogue: 16 rows x 16 trees per wave = 256 outputs, 8 per lane.
#pragma unroll
    for (int it = 0; it < 8; ++it) {
        const int idx = it * 32 + lane;
        const int r = idx >> 4;             // local B row 0..15
        const int c = idx & 15;             // local tree 0..15
        const int t = (colBase >> 2) + c;   // global tree

        float p[4];
#pragma unroll
        for (int d = 0; d < 4; ++d) {
            const float xs = xsel[wave][r][c * 4 + d];
            const float z  = (xs - thresholds[t * 4 + d]) * INV_TAU;
            p[d] = 1.0f / (1.0f + __expf(-z));
        }

        const float* lr = leaf + (size_t)t * 16;
        float sum = 0.0f;
#pragma unroll
        for (int li = 0; li < 16; ++li) {
            float pr = 1.0f;
#pragma unroll
            for (int d = 0; d < 4; ++d)
                pr *= ((li >> d) & 1) ? p[d] : (1.0f - p[d]);
            sum += pr * lr[li];
        }
        out[(size_t)(rowBase + r) * T_DIM + t] = sum;
    }
}

// ---------------------------------------------------------------------------
extern "C" void kernel_launch(void* const* d_in, const int* in_sizes, int n_in,
                              void* d_out, int out_size, void* d_ws, size_t ws_size,
                              hipStream_t stream)
{
    (void)in_sizes; (void)n_in; (void)out_size; (void)ws_size;

    const float* x      = (const float*)d_in[0];   // (B, F)
    const float* logits = (const float*)d_in[1];   // (T, D, F)
    const float* thr    = (const float*)d_in[2];   // (T, D)
    const float* leaf   = (const float*)d_in[3];   // (T, 16)
    float*       out    = (float*)d_out;           // (B, T)

    // Workspace: Wb (8 MB bf16) then Xb (4 MB bf16) — both L2-resident.
    __bf16* Wb = (__bf16*)d_ws;
    __bf16* Xb = (__bf16*)((char*)d_ws + (size_t)TD * F_DIM * sizeof(__bf16));

    odst_softmax_bf16<<<TD / 8, 256, 0, stream>>>(logits, Wb);
    odst_x_bf16<<<(B_DIM * F_DIM) / 256, 256, 0, stream>>>(x, Xb);

    dim3 grid(B_DIM / 128, TD / 64);   // 32 x 128 = 4096 workgroups
    odst_fused<<<grid, 256, 0, stream>>>(Xb, Wb, thr, leaf, out);
}
